// RRN_53764400611569
// MI455X (gfx1250) — matrix operand order repaired and checked by reading, not compiled, not run on hardware
//
#include <hip/hip_runtime.h>

typedef __attribute__((ext_vector_type(16))) _Float16 v16h;
typedef __attribute__((ext_vector_type(8)))  _Float16 v8h;
typedef __attribute__((ext_vector_type(8)))  float    v8f;

#define RRN_NB   64
#define RRN_NN   64
#define RRN_NH   96
#define RRN_BN   (RRN_NB * RRN_NN)          // 4096 rows
#define RRN_BNH  (RRN_BN * RRN_NH)          // 393216

#define T1_STR 104   // halves per row (208B, 16B aligned)
#define T3_STR 100   // floats per row

// ---------------- WMMA fragment helpers ----------------
__device__ __forceinline__ v16h make_v16(v8h lo, v8h hi) {
    return __builtin_shufflevector(lo, hi, 0,1,2,3,4,5,6,7,8,9,10,11,12,13,14,15);
}
// A 16x32 f16: lanes 0-15 hold K 0-7 & 16-23 of row M=lane; lanes 16-31 hold K 8-15 & 24-31.
__device__ __forceinline__ v16h load_a_frag(const _Float16* row, int kc, int lane) {
    const int kb = kc * 32 + ((lane >> 4) << 3);
    v8h lo = *(const v8h*)(row + kb);
    v8h hi = *(const v8h*)(row + kb + 16);
    return make_v16(lo, hi);
}
// B 32x16 f16: lanes 0-15 hold K 0-7 & 16-23 of col N=lane; lanes 16-31 hold K 8-15 & 24-31.
// Source row is W[n][*] (row-major [96][96] f16), since B[k][n] = W[n][k].
__device__ __forceinline__ v16h load_b_frag(const _Float16* row, int kc, int lane) {
    const int k0 = kc * 32 + ((lane >> 4) << 3);
    v8h lo = *(const v8h*)(row + k0);
    v8h hi = *(const v8h*)(row + k0 + 16);
    return make_v16(lo, hi);
}
__device__ __forceinline__ v8f wmma16(v16h a, v16h b, v8f c) {
    return __builtin_amdgcn_wmma_f32_16x16x32_f16(false, a, false, b, (short)0, c, false, false);
}

// ---------------- f32 -> f16 weight conversion ----------------
__global__ void cvt_f16_kernel(const float* __restrict__ src, _Float16* __restrict__ dst, int n) {
    int i = blockIdx.x * blockDim.x + threadIdx.x;
    if (i < n) dst[i] = (_Float16)src[i];
}

// ---------------- embed lookup + emb MLP; hx := xe when time_step==0 ----------------
__global__ __launch_bounds__(96) void embed_kernel(
    const int* __restrict__ x, const float* __restrict__ emb,
    const float* __restrict__ W1, const float* __restrict__ b1,
    const float* __restrict__ W2, const float* __restrict__ b2,
    const float* __restrict__ W4, const float* __restrict__ b4,
    const float* __restrict__ hx_in, const int* __restrict__ ts,
    float* __restrict__ xe, float* __restrict__ hxw)
{
    __shared__ float sin16[16], h1[96], h2[96];
    const int row = blockIdx.x, tid = threadIdx.x;
    if (tid < 16) sin16[tid] = emb[x[row] * 16 + tid];
    __syncthreads();
    float a = b1[tid];
    const float* wr = W1 + tid * 16;
    #pragma unroll
    for (int k = 0; k < 16; ++k) a += wr[k] * sin16[k];
    h1[tid] = fmaxf(a, 0.f);
    __syncthreads();
    a = b2[tid]; wr = W2 + tid * 96;
    for (int k = 0; k < 96; ++k) a += wr[k] * h1[k];
    h2[tid] = fmaxf(a, 0.f);
    __syncthreads();
    a = b4[tid]; wr = W4 + tid * 96;
    for (int k = 0; k < 96; ++k) a += wr[k] * h2[k];
    const size_t o = (size_t)row * 96 + tid;
    xe[o] = a;
    hxw[o] = (ts[0] == 0) ? a : hx_in[o];
}

// ---------------- U = W1a @ h, V = W1b @ h (msg layer-1 factorization) ----------------
__global__ __launch_bounds__(96) void uv_kernel(
    const float* __restrict__ hxw, const float* __restrict__ mW1,  // [96][192]
    float* __restrict__ U, float* __restrict__ V)
{
    __shared__ float h[96];
    const int row = blockIdx.x, tid = threadIdx.x;
    h[tid] = hxw[(size_t)row * 96 + tid];
    __syncthreads();
    float u = 0.f, v = 0.f;
    const float* wr = mW1 + tid * 192;
    for (int k = 0; k < 96; ++k) { u += wr[k] * h[k]; v += wr[96 + k] * h[k]; }
    U[(size_t)row * 96 + tid] = u;
    V[(size_t)row * 96 + tid] = v;
}

// ---------------- message MLP layers 2-3 via WMMA + masked aggregation ----------------
// One workgroup per (b,i). 128 threads = 4 waves; wave w owns rows 16w..16w+15.
__global__ __launch_bounds__(128) void msg_gemm_kernel(
    const float* __restrict__ U, const float* __restrict__ V,
    const float* __restrict__ mb1, const float* __restrict__ mb2,
    const float* __restrict__ mb4,
    const _Float16* __restrict__ W2h, const _Float16* __restrict__ W4h,  // [96][96]
    const float* __restrict__ mask, float* __restrict__ agg)
{
    __shared__ _Float16 t1h[64 * T1_STR];
    __shared__ _Float16 t2h[64 * T1_STR];
    __shared__ float    t3f[64 * T3_STR];
    __shared__ float    vb1[96];

    const int bi  = blockIdx.x;      // b*64 + i
    const int b   = bi >> 6;
    const int i   = bi & 63;
    const int tid = threadIdx.x;

    if (tid < 96) vb1[tid] = V[(size_t)bi * 96 + tid] + mb1[tid];
    __syncthreads();

    // t1[j][k] = relu(U[b,j,k] + V[b,i,k] + b1[k]) as f16
    for (int idx = tid; idx < 64 * 96; idx += 128) {
        const int j = idx / 96, k = idx - j * 96;
        float v = U[((size_t)b * 64 + j) * 96 + k] + vb1[k];
        t1h[j * T1_STR + k] = (_Float16)(v > 0.f ? v : 0.f);
    }
    __syncthreads();

    const int wave = tid >> 5;
    const int lane = tid & 31;
    const int m0   = wave * 16;
    const int mrow = m0 + (lane & 15);
    const int nlo  = lane & 15;
    const int mhi  = (lane >> 4) * 8;   // D rows: m0 + r + mhi

    // ---- GEMM1: t2 = relu(t1 @ W2^T + b2) ----
    {
        const _Float16* arow = t1h + mrow * T1_STR;
        v16h a0 = load_a_frag(arow, 0, lane);
        v16h a1 = load_a_frag(arow, 1, lane);
        v16h a2 = load_a_frag(arow, 2, lane);
        #pragma unroll
        for (int nt = 0; nt < 6; ++nt) {
            const int n = nt * 16 + nlo;
            const _Float16* wrow = W2h + n * 96;
            v8f acc = {};
            acc = wmma16(a0, load_b_frag(wrow, 0, lane), acc);
            acc = wmma16(a1, load_b_frag(wrow, 1, lane), acc);
            acc = wmma16(a2, load_b_frag(wrow, 2, lane), acc);
            const float bias = mb2[n];
            #pragma unroll
            for (int r = 0; r < 8; ++r) {
                const int m = m0 + r + mhi;
                float v = acc[r] + bias;
                t2h[m * T1_STR + n] = (_Float16)(v > 0.f ? v : 0.f);
            }
        }
    }
    __syncthreads();

    // ---- GEMM2: t3 = t2 @ W4^T + b4 ----
    {
        const _Float16* arow = t2h + mrow * T1_STR;
        v16h a0 = load_a_frag(arow, 0, lane);
        v16h a1 = load_a_frag(arow, 1, lane);
        v16h a2 = load_a_frag(arow, 2, lane);
        #pragma unroll
        for (int nt = 0; nt < 6; ++nt) {
            const int n = nt * 16 + nlo;
            const _Float16* wrow = W4h + n * 96;
            v8f acc = {};
            acc = wmma16(a0, load_b_frag(wrow, 0, lane), acc);
            acc = wmma16(a1, load_b_frag(wrow, 1, lane), acc);
            acc = wmma16(a2, load_b_frag(wrow, 2, lane), acc);
            const float bias = mb4[n];
            #pragma unroll
            for (int r = 0; r < 8; ++r) {
                const int m = m0 + r + mhi;
                t3f[m * T3_STR + n] = acc[r] + bias;
            }
        }
    }
    __syncthreads();

    // ---- agg[b,i,h] = sum_j mask[i,j] * t3[j,h] ----
    if (tid < 96) {
        const float* mr = mask + i * 64;
        float s = 0.f;
        for (int j = 0; j < 64; ++j) s += mr[j] * t3f[j * T3_STR + tid];
        agg[(size_t)bi * 96 + tid] = s;
    }
}

// ---------------- node MLP on concat(xe, agg) ----------------
__global__ __launch_bounds__(96) void node_kernel(
    const float* __restrict__ xe, const float* __restrict__ agg,
    const float* __restrict__ W1, const float* __restrict__ b1,
    const float* __restrict__ W2, const float* __restrict__ b2,
    const float* __restrict__ W4, const float* __restrict__ b4,
    float* __restrict__ nodein)
{
    __shared__ float in0[192], h1[96], h2[96];
    const int row = blockIdx.x, tid = threadIdx.x;
    in0[tid]      = xe[(size_t)row * 96 + tid];
    in0[96 + tid] = agg[(size_t)row * 96 + tid];
    __syncthreads();
    float a = b1[tid];
    const float* wr = W1 + tid * 192;
    for (int k = 0; k < 192; ++k) a += wr[k] * in0[k];
    h1[tid] = fmaxf(a, 0.f);
    __syncthreads();
    a = b2[tid]; wr = W2 + tid * 96;
    for (int k = 0; k < 96; ++k) a += wr[k] * h1[k];
    h2[tid] = fmaxf(a, 0.f);
    __syncthreads();
    a = b4[tid]; wr = W4 + tid * 96;
    for (int k = 0; k < 96; ++k) a += wr[k] * h2[k];
    nodein[(size_t)row * 96 + tid] = a;
}

// ---------------- LSTM gate pre-activations ----------------
__global__ __launch_bounds__(128) void gates_kernel(
    const float* __restrict__ nodein, const float* __restrict__ hxw,
    const float* __restrict__ W_ih, const float* __restrict__ W_hh,
    const float* __restrict__ b_ih, const float* __restrict__ b_hh,
    float* __restrict__ gates)
{
    __shared__ float nr[96], hr[96];
    const int row = blockIdx.x, tid = threadIdx.x;
    if (tid < 96) {
        nr[tid] = nodein[(size_t)row * 96 + tid];
        hr[tid] = hxw[(size_t)row * 96 + tid];
    }
    __syncthreads();
    for (int g = tid; g < 384; g += 128) {
        float a = b_ih[g] + b_hh[g];
        const float* wi = W_ih + g * 96;
        const float* wh = W_hh + g * 96;
        for (int k = 0; k < 96; ++k) a += wi[k] * nr[k] + wh[k] * hr[k];
        gates[(size_t)row * 384 + g] = a;
    }
}

// ---------------- LSTM elementwise + output head + softmax ----------------
__global__ __launch_bounds__(96) void lstm_out_kernel(
    const float* __restrict__ gates, const float* __restrict__ sx,
    const float* __restrict__ W_out, const float* __restrict__ b_out,
    float* __restrict__ out)  // [ht | st | pred]
{
    __shared__ float hts[96], lg[8];
    const int row = blockIdx.x, tid = threadIdx.x;
    const float* g = gates + (size_t)row * 384;
    const float ig = g[tid], fg = g[96 + tid], gg = g[192 + tid], og = g[288 + tid];
    const float si = 1.f / (1.f + __expf(-ig));
    const float sf = 1.f / (1.f + __expf(-fg));
    const float so = 1.f / (1.f + __expf(-og));
    const float st = sf * sx[(size_t)row * 96 + tid] + si * tanhf(gg);
    const float ht = so * tanhf(st);
    out[(size_t)row * 96 + tid] = ht;
    out[(size_t)RRN_BNH + (size_t)row * 96 + tid] = st;
    hts[tid] = ht;
    __syncthreads();
    if (tid < 8) {
        float a = b_out[tid];
        const float* wr = W_out + tid * 96;
        for (int k = 0; k < 96; ++k) a += wr[k] * hts[k];
        lg[tid] = a;
    }
    __syncthreads();
    if (tid < 8) {
        float m = lg[0];
        for (int t = 1; t < 8; ++t) m = fmaxf(m, lg[t]);
        float s = 0.f;
        for (int t = 0; t < 8; ++t) s += __expf(lg[t] - m);
        out[2 * (size_t)RRN_BNH + (size_t)row * 8 + tid] = __expf(lg[tid] - m) / s;
    }
}

extern "C" void kernel_launch(void* const* d_in, const int* in_sizes, int n_in,
                              void* d_out, int out_size, void* d_ws, size_t ws_size,
                              hipStream_t stream) {
    (void)in_sizes; (void)n_in; (void)out_size; (void)ws_size;
    // setup_inputs() flattened order:
    const int*   x        = (const int*)  d_in[0];
    // d_in[1]=r, d_in[2]=c are discarded by the reference
    const float* hx_in    = (const float*)d_in[3];
    const float* sx       = (const float*)d_in[4];
    const int*   ts       = (const int*)  d_in[5];
    const float* mask     = (const float*)d_in[6];
    const float* emb      = (const float*)d_in[7];
    const float* eW1 = (const float*)d_in[8],  *eb1 = (const float*)d_in[9];
    const float* eW2 = (const float*)d_in[10], *eb2 = (const float*)d_in[11];
    const float* eW4 = (const float*)d_in[12], *eb4 = (const float*)d_in[13];
    const float* mW1 = (const float*)d_in[14], *mb1 = (const float*)d_in[15];
    const float* mW2 = (const float*)d_in[16], *mb2 = (const float*)d_in[17];
    const float* mW4 = (const float*)d_in[18], *mb4 = (const float*)d_in[19];
    const float* nW1 = (const float*)d_in[20], *nb1 = (const float*)d_in[21];
    const float* nW2 = (const float*)d_in[22], *nb2 = (const float*)d_in[23];
    const float* nW4 = (const float*)d_in[24], *nb4 = (const float*)d_in[25];
    const float* W_ih = (const float*)d_in[26];
    const float* W_hh = (const float*)d_in[27];
    const float* b_ih = (const float*)d_in[28];
    const float* b_hh = (const float*)d_in[29];
    const float* W_out = (const float*)d_in[30];
    const float* b_out = (const float*)d_in[31];
    float* out = (float*)d_out;

    // workspace layout (bytes)
    char* ws = (char*)d_ws;
    const size_t SZ = (size_t)RRN_BNH * 4;              // 1,572,864 B
    float*    xe     = (float*)(ws);
    float*    hxw    = (float*)(ws + 1 * SZ);
    float*    U      = (float*)(ws + 2 * SZ);
    float*    V      = (float*)(ws + 3 * SZ);
    float*    agg    = (float*)(ws + 4 * SZ);
    float*    nodein = (float*)(ws + 5 * SZ);
    float*    gates  = (float*)(ws + 6 * SZ);           // 4096*384*4 = 6,291,456 B
    _Float16* W2h    = (_Float16*)(ws + 6 * SZ + (size_t)RRN_BN * 384 * 4);
    _Float16* W4h    = (_Float16*)((char*)W2h + 96 * 96 * sizeof(_Float16));

    // 0) convert message-MLP GEMM weights to f16 (L2-resident, reused by all 4096 WGs)
    cvt_f16_kernel<<<36, 256, 0, stream>>>(mW2, W2h, 96 * 96);
    cvt_f16_kernel<<<36, 256, 0, stream>>>(mW4, W4h, 96 * 96);

    // 1) embedding MLP; hx := xe (time_step==0 path)
    embed_kernel<<<RRN_BN, 96, 0, stream>>>(x, emb, eW1, eb1, eW2, eb2, eW4, eb4,
                                            hx_in, ts, xe, hxw);
    // 2) factorized message layer-1: U = W1[:, :96]@h, V = W1[:, 96:]@h
    uv_kernel<<<RRN_BN, 96, 0, stream>>>(hxw, mW1, U, V);
    // 3) dominant compute: per-(b,i) 64x96x96 WMMA GEMMs + masked aggregation
    msg_gemm_kernel<<<RRN_BN, 128, 0, stream>>>(U, V, mb1, mb2, mb4, W2h, W4h, mask, agg);
    // 4) node MLP
    node_kernel<<<RRN_BN, 96, 0, stream>>>(xe, agg, nW1, nb1, nW2, nb2, nW4, nb4, nodein);
    // 5) LSTM gates
    gates_kernel<<<RRN_BN, 128, 0, stream>>>(nodein, hxw, W_ih, W_hh, b_ih, b_hh, gates);
    // 6) LSTM elementwise + output projection + softmax
    lstm_out_kernel<<<RRN_BN, 96, 0, stream>>>(gates, sx, W_out, b_out, out);
}
